// AttentionPool1d_48850958025248
// MI455X (gfx1250) — compile-verified, matrix operand order checked by reading, not run on hardware
//
#include <hip/hip_runtime.h>
#include <stdint.h>

// ---------------------------------------------------------------------------
// AttentionPool1d on MI455X (gfx1250)
//   B=16, D=512, T=2048, window=9, pad=4, TEMP=1.5
//   Two f16-WMMA GEMMs (f32 accum) + fp32 window softmax.
//   Weights pre-converted to fp16 once; B tiles staged by the Tensor Data
//   Mover (tensor_load_to_lds) when available, double-buffered LDS pipeline.
// ---------------------------------------------------------------------------

typedef __attribute__((ext_vector_type(16))) _Float16 v16h;
typedef __attribute__((ext_vector_type(8)))  _Float16 v8h;
typedef __attribute__((ext_vector_type(4)))  _Float16 v4h;
typedef __attribute__((ext_vector_type(8)))  float    v8f;
typedef __attribute__((ext_vector_type(4)))  float    v4f;
typedef __attribute__((ext_vector_type(4)))  unsigned int v4u;
typedef __attribute__((ext_vector_type(8)))  int      v8i;
typedef __attribute__((ext_vector_type(4)))  int      v4i;

#define DIM   512
#define TLEN  2048
#define BATCH 16
#define KW    9
#define KPAD  4

#if defined(__HIP_DEVICE_COMPILE__) && defined(__gfx1250__) && \
    __has_builtin(__builtin_amdgcn_tensor_load_to_lds) && \
    __has_builtin(__builtin_amdgcn_s_wait_tensorcnt)
#define HAVE_TDM 1
#else
#define HAVE_TDM 0
#endif

// ---------------------------------------------------------------------------
// One-time weight conversion: fp32 (512x512) -> fp16 (512x512)
// ---------------------------------------------------------------------------
__global__ __launch_bounds__(256)
void wcvt_kernel(const float* __restrict__ Wf, _Float16* __restrict__ Wh)
{
    const int i = (blockIdx.x * 256 + threadIdx.x) * 4;
    v4f v = *(const v4f*)(Wf + i);
    v4h h;
    h[0] = (_Float16)v[0]; h[1] = (_Float16)v[1];
    h[2] = (_Float16)v[2]; h[3] = (_Float16)v[3];
    *(v4h*)(Wh + i) = h;
}

// ---------------------------------------------------------------------------
// Tiled WMMA GEMM:  Out[b, m, n] = A[b, :, m] . Wh[n, :] + bias[n]
//   A layout (B, K=512, M=2048), M (=t) contiguous  (x fp32 or pooled fp16)
//   Wh is (N=512, K=512) row-major fp16 (pre-converted)
// Block: 256 thr / 8 waves. Block tile 128(m) x 128(n). Wave tile 32x64.
// Double-buffered LDS; B tile staged via TDM (wave 0) when available.
// ---------------------------------------------------------------------------
template<bool A_IS_F32, bool OUT_TRANS>
__global__ __launch_bounds__(256)
void gemm_wmma_kernel(const void* __restrict__ Aglob,
                      const _Float16* __restrict__ Wh,
                      const float* __restrict__ bias,
                      float* __restrict__ Out)
{
    constexpr int D = DIM, T = TLEN;
    __shared__ __align__(16) _Float16 Ash[2][128][40];  // [t][k], 32->40 pad
    __shared__ __align__(16) _Float16 Bsh[2][128][40];  // [n][k], 32->40 pad

    const int tid  = threadIdx.x;
    const int lane = tid & 31;
    const int wave = tid >> 5;
    const int b    = blockIdx.z;
    const int t0   = blockIdx.y * 128;
    const int n0   = blockIdx.x * 128;

    const int m0w = (wave & 3) * 32;   // wave m-offset inside block tile
    const int n0w = (wave >> 2) * 64;  // wave n-offset inside block tile

    v8f acc[2][4];
#pragma unroll
    for (int mi = 0; mi < 2; ++mi)
#pragma unroll
        for (int ni = 0; ni < 4; ++ni)
#pragma unroll
            for (int r = 0; r < 8; ++r) acc[mi][ni][r] = 0.0f;

    const float*    Af32 = (const float*)Aglob    + (size_t)b * D * T;
    const _Float16* Af16 = (const _Float16*)Aglob + (size_t)b * D * T;

    const int kb = (lane < 16) ? 0 : 8;   // K-half select, 16-bit WMMA layout
    const int mr = lane & 15;

    // ---- A tile: 32(k) x 128(t), transpose into Ash[buf][t][k] ----
    auto stageA = [&](int k0, int buf) {
#pragma unroll
        for (int i = 0; i < 4; ++i) {
            const int j  = i * 256 + tid;   // 1024 chunks of 4 elements
            const int kl = j >> 5;          // 0..31
            const int tl = (j & 31) << 2;   // 0..124
            if (A_IS_F32) {
                v4f v = *(const v4f*)(Af32 + (size_t)(k0 + kl) * T + t0 + tl);
                Ash[buf][tl + 0][kl] = (_Float16)v[0];
                Ash[buf][tl + 1][kl] = (_Float16)v[1];
                Ash[buf][tl + 2][kl] = (_Float16)v[2];
                Ash[buf][tl + 3][kl] = (_Float16)v[3];
            } else {
                v4h v = *(const v4h*)(Af16 + (size_t)(k0 + kl) * T + t0 + tl);
                Ash[buf][tl + 0][kl] = v[0];
                Ash[buf][tl + 1][kl] = v[1];
                Ash[buf][tl + 2][kl] = v[2];
                Ash[buf][tl + 3][kl] = v[3];
            }
        }
    };

#if HAVE_TDM
    // ---- B tile via Tensor Data Mover: 128(n) rows x 32(k) halves,
    //      global row stride 512 halves; LDS padded 64B data + 16B pad -> 80B
    auto stageB = [&](int k0, int buf) {
        if (wave == 0) {
            const unsigned long long ga =
                (unsigned long long)(uintptr_t)(Wh + (size_t)n0 * D + k0);
            const unsigned int lds = (unsigned int)(uintptr_t)&Bsh[buf][0][0];
            v4u g0;
            g0[0] = 1u;                                   // count=1 (valid)
            g0[1] = lds;                                  // lds_addr
            g0[2] = (unsigned int)(ga & 0xffffffffu);     // global_addr lo
            g0[3] = (unsigned int)((ga >> 32) & 0x01ffffffu) | (2u << 30); // type=2
            v8i g1;
            g1[0] = (1 << 16)    // data_size = 2 bytes
                  | (1 << 20)    // pad_enable
                  | (3 << 22)    // pad_interval: 16 DWORDs (= 64B row)
                  | (3 << 25);   // pad_amount:    4 DWORDs (= 16B pad)
            g1[1] = (int)(((unsigned)D & 0xffffu) << 16); // tensor_dim0 lo16
            g1[2] = (int)(((unsigned)D & 0xffffu) << 16); // tensor_dim1 lo16
            g1[3] = (int)(32u << 16);                     // tile_dim0 = 32 (k)
            g1[4] = 128;                                  // tile_dim1 = 128 (n)
            g1[5] = D;                                    // tensor_dim0_stride
            g1[6] = 0;
            g1[7] = 0;
            v4i gz; gz[0] = 0; gz[1] = 0; gz[2] = 0; gz[3] = 0;
#if defined(__clang_major__) && __clang_major__ >= 23
            v8i gz8;
#pragma unroll
            for (int q = 0; q < 8; ++q) gz8[q] = 0;
            __builtin_amdgcn_tensor_load_to_lds(g0, g1, gz, gz, gz8, 0);
#else
            __builtin_amdgcn_tensor_load_to_lds(g0, g1, gz, gz, 0);
#endif
        }
    };
#else
    // ---- B tile fallback: plain fp16 copy 128(n) x 32(k) ----
    auto stageB = [&](int k0, int buf) {
#pragma unroll
        for (int i = 0; i < 2; ++i) {
            const int j  = i * 256 + tid;   // 512 chunks of 8 halves
            const int nl = j >> 2;          // 0..127
            const int kl = (j & 3) << 3;    // 0,8,16,24
            *(v8h*)&Bsh[buf][nl][kl] =
                *(const v8h*)(Wh + (size_t)(n0 + nl) * D + k0 + kl);
        }
    };
#endif

    auto compute = [&](int buf) {
        v16h af[2], bf[4];
#pragma unroll
        for (int mi = 0; mi < 2; ++mi) {
            const _Float16* base = &Ash[buf][m0w + mi * 16 + mr][0];
            ((v8h*)&af[mi])[0] = *(const v8h*)(base + kb);
            ((v8h*)&af[mi])[1] = *(const v8h*)(base + kb + 16);
        }
#pragma unroll
        for (int ni = 0; ni < 4; ++ni) {
            const _Float16* base = &Bsh[buf][n0w + ni * 16 + mr][0];
            ((v8h*)&bf[ni])[0] = *(const v8h*)(base + kb);
            ((v8h*)&bf[ni])[1] = *(const v8h*)(base + kb + 16);
        }
#pragma unroll
        for (int mi = 0; mi < 2; ++mi)
#pragma unroll
            for (int ni = 0; ni < 4; ++ni)
                acc[mi][ni] = __builtin_amdgcn_wmma_f32_16x16x32_f16(
                    false, af[mi], false, bf[ni],
                    (short)0, acc[mi][ni], false, false);
    };

    // ---- software pipeline over K (16 steps of 32), double-buffered ----
    stageA(0, 0);
    stageB(0, 0);
#if HAVE_TDM
    if (wave == 0) __builtin_amdgcn_s_wait_tensorcnt(0);
#endif
    __syncthreads();

    for (int ks = 0; ks < D / 32; ++ks) {
        const int cur = ks & 1;
        const int nxt = cur ^ 1;
        if (ks + 1 < D / 32) {
            stageA((ks + 1) * 32, nxt);
            stageB((ks + 1) * 32, nxt);
        }
        compute(cur);
#if HAVE_TDM
        if (ks + 1 < D / 32 && wave == 0) __builtin_amdgcn_s_wait_tensorcnt(0);
#endif
        __syncthreads();
    }

    // ---- epilogue: bias + store (C layout: M = r + 8*(lane>=16), N = lane%16)
    const int rhi = (lane >> 4) << 3;
#pragma unroll
    for (int mi = 0; mi < 2; ++mi)
#pragma unroll
        for (int ni = 0; ni < 4; ++ni) {
            const int dcol = n0 + n0w + ni * 16 + (lane & 15);
            const float bv = bias[dcol];
#pragma unroll
            for (int r = 0; r < 8; ++r) {
                const int trow = t0 + m0w + mi * 16 + r + rhi;
                const float val = acc[mi][ni][r] + bv;
                if (OUT_TRANS)
                    Out[((size_t)b * D + dcol) * T + trow] = val;   // [b][d][t]
                else
                    Out[((size_t)b * T + trow) * D + dcol] = val;   // [b][t][d]
            }
        }
}

// ---------------------------------------------------------------------------
// Window attention: energies from Y, softmax, weighted pool of raw x.
// Block: 256 thr handles (b, 32 consecutive t). pooled written (B, D, T) f16.
// ---------------------------------------------------------------------------
__global__ __launch_bounds__(256)
void attn_pool_kernel(const float* __restrict__ Y,      // [B][T][D]
                      const float* __restrict__ x,      // [B][D][T]
                      const float* __restrict__ bq,     // [D]
                      _Float16* __restrict__ pooled)    // [B][D][T]
{
    constexpr int D = DIM, T = TLEN;
    __shared__ float attn_s[32][KW];

    const int tid  = threadIdx.x;
    const int lane = tid & 31;
    const int wave = tid >> 5;
    const int b    = blockIdx.x >> 6;          // 64 blocks per batch
    const int t0   = (blockIdx.x & 63) * 32;

    const float inv = 1.0f / (22.627417f * 1.5f);   // 1/(sqrt(D)*TEMP)

    for (int q = 0; q < 4; ++q) {
        const int tl = wave * 4 + q;
        const int t  = t0 + tl;

        const float* yc = Y + ((size_t)b * T + t) * D + lane * 16;
        float c[16];
#pragma unroll
        for (int i = 0; i < 16; i += 4) {
            v4f v = *(const v4f*)(yc + i);
            c[i] = v[0]; c[i + 1] = v[1]; c[i + 2] = v[2]; c[i + 3] = v[3];
        }

        float e[KW];
#pragma unroll
        for (int w = 0; w < KW; ++w) {
            const int s = t + w - KPAD;
            const float* yn = (s >= 0 && s < T)
                                  ? Y + ((size_t)b * T + s) * D + lane * 16
                                  : bq + lane * 16;   // padded window -> K = bq
            float a = 0.0f;
#pragma unroll
            for (int i = 0; i < 16; i += 4) {
                v4f v = *(const v4f*)(yn + i);
                a += c[i] * v[0] + c[i + 1] * v[1] + c[i + 2] * v[2] + c[i + 3] * v[3];
            }
            e[w] = a;
        }
#pragma unroll
        for (int w = 0; w < KW; ++w)
#pragma unroll
            for (int off = 16; off > 0; off >>= 1)
                e[w] += __shfl_xor(e[w], off, 32);

        float m = -1e30f;
#pragma unroll
        for (int w = 0; w < KW; ++w) { e[w] *= inv; m = fmaxf(m, e[w]); }
        float sum = 0.0f;
#pragma unroll
        for (int w = 0; w < KW; ++w) { e[w] = __expf(e[w] - m); sum += e[w]; }
        const float rs = 1.0f / sum;
        if (lane == 0) {
#pragma unroll
            for (int w = 0; w < KW; ++w) attn_s[tl][w] = e[w] * rs;
        }
    }
    __syncthreads();

    const int t_idx = tid & 31;     // lanes walk contiguous t -> coalesced
    const int t     = t0 + t_idx;
    float a[KW];
#pragma unroll
    for (int w = 0; w < KW; ++w) a[w] = attn_s[t_idx][w];

    for (int d = tid >> 5; d < D; d += 8) {
        const float* xr = x + ((size_t)b * D + d) * T + t;
        float accp = 0.0f;
#pragma unroll
        for (int w = 0; w < KW; ++w) {
            const int s = t + w - KPAD;
            const float xv = (s >= 0 && s < T) ? xr[w - KPAD] : 0.0f;
            accp += a[w] * xv;
        }
        pooled[((size_t)b * D + d) * T + t] = (_Float16)accp;
    }
}

// ---------------------------------------------------------------------------
extern "C" void kernel_launch(void* const* d_in, const int* in_sizes, int n_in,
                              void* d_out, int out_size, void* d_ws, size_t ws_size,
                              hipStream_t stream)
{
    const float* x  = (const float*)d_in[0];   // (B, D, T) fp32
    const float* Wq = (const float*)d_in[1];   // (D, D)
    const float* bq = (const float*)d_in[2];   // (D,)
    const float* Wo = (const float*)d_in[3];   // (D, D)
    const float* bo = (const float*)d_in[4];   // (D,)
    float* out = (float*)d_out;                // (B, D, T) fp32

    char* ws = (char*)d_ws;
    float*    Y      = (float*)ws;                                   // 64 MB
    _Float16* pooled = (_Float16*)(ws + (size_t)BATCH * TLEN * DIM * 4);  // +32 MB
    _Float16* Whq    = (_Float16*)(ws + (size_t)BATCH * TLEN * DIM * 6);  // +0.5 MB
    _Float16* Who    = Whq + (size_t)DIM * DIM;                          // +0.5 MB

    dim3 block(256);

    // 0) one-time fp16 copies of the weights
    wcvt_kernel<<<dim3(DIM * DIM / 1024), block, 0, stream>>>(Wq, Whq);
    wcvt_kernel<<<dim3(DIM * DIM / 1024), block, 0, stream>>>(Wo, Who);

    dim3 gridG(DIM / 128, TLEN / 128, BATCH);   // (4, 16, 16)

    // 1) Y[b,t,:] = x[b,:,t] . Wq^T + bq   (f16 WMMA, f32 accum)
    gemm_wmma_kernel<true, false><<<gridG, block, 0, stream>>>(x, Whq, bq, Y);

    // 2) window energies -> softmax -> pooled (B,D,T) fp16
    attn_pool_kernel<<<dim3(BATCH * (TLEN / 32)), block, 0, stream>>>(Y, x, bq, pooled);

    // 3) out[b,:,t] = pooled[b,:,t] . Wo^T + bo, transposed into d_out
    gemm_wmma_kernel<false, true><<<gridG, block, 0, stream>>>(pooled, Who, bo, out);
}
